// DigitConvolutionalModel_19481971655263
// MI455X (gfx1250) — matrix operand order repaired and checked by reading, not compiled, not run on hardware
//
#include <hip/hip_runtime.h>

// ---------------------------------------------------------------------------
// Fused DigitConvolutionalModel for gfx1250 (MI455X), bf16 WMMA pipeline.
//   conv3x3 folded into W1  =>  x[ B x 784 ] @ W1eff[784 x 128] (+b1, relu)
//   -> 2x (h @ W2 + b2, relu)  -> h @ W3 + b3
// All GEMMs via v_wmma_f32_16x16x32_bf16; x read exactly once from HBM.
// ---------------------------------------------------------------------------

typedef __attribute__((ext_vector_type(16))) __bf16 v16bf;
typedef __attribute__((ext_vector_type(8)))  float  v8f;

#define MT  64          // batch rows per block
#define XAS 808         // LDS stride for padded x tile (784 -> 800 used, 808 stored)
#define HS  136         // LDS stride for 128-wide hidden activations

union FragBF { uint4 q[2]; v16bf v; };

__device__ __forceinline__ unsigned short f2bfbits(float f) {
    union { float f; unsigned u; } v; v.f = f;
    unsigned r = v.u + 0x7FFFu + ((v.u >> 16) & 1u);   // round-to-nearest-even
    return (unsigned short)(r >> 16);
}
__device__ __forceinline__ __bf16 f2bf(float f) {
    union { unsigned short u; __bf16 b; } o; o.u = f2bfbits(f); return o.b;
}

__device__ __forceinline__ v8f wmma_bf16(const FragBF& a, const FragBF& b, v8f c) {
    return __builtin_amdgcn_wmma_f32_16x16x32_bf16(
        /*neg_a=*/false, a.v, /*neg_b=*/false, b.v,
        /*c_mod=*/(short)0, c, /*reuse_a=*/false, /*reuse_b=*/false);
}

// ---------------------------------------------------------------------------
// Prep: build W1eff^T (conv folded, [128][800] bf16), W2^T ([128][128] bf16),
// W3^T zero-padded to 16 cols ([16][128] bf16). Runs fresh every launch.
// ---------------------------------------------------------------------------
extern "C" __global__ void __launch_bounds__(256)
prep_weights(const float* __restrict__ cw,  const float* __restrict__ W1,
             const float* __restrict__ W2,  const float* __restrict__ W3,
             __bf16* __restrict__ w1t, __bf16* __restrict__ w2t,
             __bf16* __restrict__ w3t)
{
    int idx = blockIdx.x * 256 + threadIdx.x;
    const int N1 = 128 * 800;
    if (idx < N1) {
        const int n = idx / 800, p = idx - n * 800;
        float acc = 0.f;
        if (p < 784) {
            const int r = p / 28, c = p - r * 28;
            #pragma unroll
            for (int ki = 0; ki < 3; ++ki) {
                const int i = r - ki;
                if (i < 0 || i >= 26) continue;
                #pragma unroll
                for (int kj = 0; kj < 3; ++kj) {
                    const int j = c - kj;
                    if (j < 0 || j >= 26) continue;
                    acc += cw[ki * 3 + kj] * W1[(i * 26 + j) * 128 + n];
                }
            }
        }
        w1t[idx] = f2bf(acc);
        return;
    }
    idx -= N1;
    if (idx < 128 * 128) {
        const int n = idx >> 7, k = idx & 127;
        w2t[idx] = f2bf(W2[k * 128 + n]);
        return;
    }
    idx -= 128 * 128;
    if (idx < 16 * 128) {
        const int n = idx >> 7, k = idx & 127;
        w3t[idx] = f2bf(n < 10 ? W3[k * 10 + n] : 0.f);
    }
}

// ---------------------------------------------------------------------------
// One 128-wide hidden layer: hdst = relu(hsrc @ W^T + bias), per-wave
// 16-row strip x 4 column tiles, K = 128 (4 k-tiles, fully unrolled).
// B-fragments for all 4 column tiles are loaded before the 4 WMMAs so their
// latencies overlap; full unroll keeps accumulators pinned (no copy/rotate).
// ---------------------------------------------------------------------------
__device__ __forceinline__ void layer128(const __bf16* hsrc, __bf16* hdst,
                                         const __bf16* __restrict__ wT,
                                         const float* __restrict__ bias,
                                         int wrow, int wcol, int lo, int hi)
{
    v8f acc[4] = {};
    const __bf16* ap_base = hsrc + (wrow + lo) * HS + hi * 8;
    const __bf16* bp_base = wT + (wcol + lo) * 128 + hi * 8;
    #pragma unroll
    for (int kt = 0; kt < 4; ++kt) {
        const int k0 = kt * 32;
        FragBF a;
        a.q[0] = *(const uint4*)(ap_base + k0);
        a.q[1] = *(const uint4*)(ap_base + k0 + 16);
        FragBF bb[4];
        #pragma unroll
        for (int t = 0; t < 4; ++t) {
            const __bf16* bp = bp_base + t * (16 * 128) + k0;
            bb[t].q[0] = *(const uint4*)bp;
            bb[t].q[1] = *(const uint4*)(bp + 16);
        }
        #pragma unroll
        for (int t = 0; t < 4; ++t)
            acc[t] = wmma_bf16(a, bb[t], acc[t]);
    }
    #pragma unroll
    for (int t = 0; t < 4; ++t) {
        const int n  = wcol + t * 16 + lo;
        const float bv = bias[n];
        #pragma unroll
        for (int r = 0; r < 8; ++r) {
            float v = acc[t][r] + bv;
            v = v > 0.f ? v : 0.f;
            hdst[(wrow + r + hi * 8) * HS + n] = f2bf(v);
        }
    }
}

// ---------------------------------------------------------------------------
// Fused network kernel. 256 threads = 8 wave32; 64 batch rows per block.
// Dynamic LDS: x tile [64][808] bf16 + two h buffers [64][136] bf16 = 135 KB.
// ---------------------------------------------------------------------------
extern "C" __global__ void __launch_bounds__(256)
fused_net(const float* __restrict__ x,
          const float* __restrict__ b1v,
          const float* __restrict__ b2v,
          const float* __restrict__ b3v,
          const __bf16* __restrict__ w1t,
          const __bf16* __restrict__ w2t,
          const __bf16* __restrict__ w3t,
          float* __restrict__ out)
{
    extern __shared__ __bf16 smem[];
    __bf16* xa = smem;                    // [MT][XAS]
    __bf16* h0 = smem + MT * XAS;         // [MT][HS]
    __bf16* h1 = h0 + MT * HS;            // [MT][HS]

    const int tid  = threadIdx.x;
    const int wave = tid >> 5;
    const int lane = tid & 31;
    const int lo   = lane & 15;
    const int hi   = lane >> 4;
    const long long b0 = (long long)blockIdx.x * MT;

    // ---- Stage 0: x tile (fp32) -> bf16 LDS, zero pad K 784..808 ----------
    {
        const int CPR = XAS / 4;          // 202 four-element chunks per row
        for (int c = tid; c < MT * CPR; c += 256) {
            const int row  = c / CPR;
            const int col4 = (c - row * CPR) * 4;
            unsigned w01 = 0u, w23 = 0u;
            if (col4 < 784) {
                const float4 f = *(const float4*)(x + (b0 + row) * 784 + col4);
                w01 = (unsigned)f2bfbits(f.x) | ((unsigned)f2bfbits(f.y) << 16);
                w23 = (unsigned)f2bfbits(f.z) | ((unsigned)f2bfbits(f.w) << 16);
            }
            *(uint2*)(xa + row * XAS + col4) = make_uint2(w01, w23);
        }
    }
    __syncthreads();

    const int wrow = (wave & 3) * 16;     // 16-row strip within tile
    const int wcol = (wave >> 2) * 64;    // 4 column tiles of 16

    // ---- Stage 1: h0 = relu(x @ W1eff + b1), K = 800 (zero-padded) --------
    // Fully unrolled: constant k-offsets fold into load immediates, all four
    // column tiles' B-frags issue as one load clause per k-tile.
    {
        v8f acc[4] = {};
        const __bf16* ap_base = xa + (wrow + lo) * XAS + hi * 8;
        const __bf16* bp_base = w1t + (wcol + lo) * 800 + hi * 8;
        #pragma unroll
        for (int kt = 0; kt < 25; ++kt) {
            const int k0 = kt * 32;
            FragBF a;
            a.q[0] = *(const uint4*)(ap_base + k0);
            a.q[1] = *(const uint4*)(ap_base + k0 + 16);
            FragBF bb[4];
            #pragma unroll
            for (int t = 0; t < 4; ++t) {
                const __bf16* bp = bp_base + t * (16 * 800) + k0;
                bb[t].q[0] = *(const uint4*)bp;
                bb[t].q[1] = *(const uint4*)(bp + 16);
            }
            #pragma unroll
            for (int t = 0; t < 4; ++t)
                acc[t] = wmma_bf16(a, bb[t], acc[t]);
        }
        #pragma unroll
        for (int t = 0; t < 4; ++t) {
            const int n  = wcol + t * 16 + lo;
            const float bv = b1v[n];
            #pragma unroll
            for (int r = 0; r < 8; ++r) {
                float v = acc[t][r] + bv;
                v = v > 0.f ? v : 0.f;
                h0[(wrow + r + hi * 8) * HS + n] = f2bf(v);
            }
        }
    }
    __syncthreads();

    // ---- Stage 2: two shared-weight mid layers (ping-pong) ----------------
    layer128(h0, h1, w2t, b2v, wrow, wcol, lo, hi);
    __syncthreads();
    layer128(h1, h0, w2t, b2v, wrow, wcol, lo, hi);
    __syncthreads();

    // ---- Stage 3: out = h @ W3 + b3 (waves 0..3, one 16x16 tile each) -----
    if (wave < 4) {                        // wave-uniform: EXEC stays all-ones
        const int m0 = wave * 16;
        v8f acc = {};
        const __bf16* ap_base = h0 + (m0 + lo) * HS + hi * 8;
        const __bf16* bp_base = w3t + lo * 128 + hi * 8;
        #pragma unroll
        for (int kt = 0; kt < 4; ++kt) {
            const int k0 = kt * 32;
            FragBF a;
            a.q[0] = *(const uint4*)(ap_base + k0);
            a.q[1] = *(const uint4*)(ap_base + k0 + 16);
            FragBF bb;
            bb.q[0] = *(const uint4*)(bp_base + k0);
            bb.q[1] = *(const uint4*)(bp_base + k0 + 16);
            acc = wmma_bf16(a, bb, acc);
        }
        if (lo < 10) {
            const float bv = b3v[lo];
            #pragma unroll
            for (int r = 0; r < 8; ++r)
                out[(b0 + m0 + r + hi * 8) * 10 + lo] = acc[r] + bv;
        }
    }
}

// ---------------------------------------------------------------------------
extern "C" void kernel_launch(void* const* d_in, const int* in_sizes, int n_in,
                              void* d_out, int out_size, void* d_ws, size_t ws_size,
                              hipStream_t stream)
{
    const float* x     = (const float*)d_in[0];
    const float* convw = (const float*)d_in[1];
    const float* W1    = (const float*)d_in[2];
    const float* b1    = (const float*)d_in[3];
    const float* W2    = (const float*)d_in[4];
    const float* b2    = (const float*)d_in[5];
    const float* W3    = (const float*)d_in[6];
    const float* b3    = (const float*)d_in[7];
    float* out = (float*)d_out;

    // Workspace: bf16 transposed/folded weights (~240 KB)
    __bf16* w1t = (__bf16*)d_ws;            // 128 x 800
    __bf16* w2t = w1t + 128 * 800;          // 128 x 128
    __bf16* w3t = w2t + 128 * 128;          // 16  x 128

    const int prep_total = 128 * 800 + 128 * 128 + 16 * 128;   // 120832
    prep_weights<<<(prep_total + 255) / 256, 256, 0, stream>>>(
        convw, W1, W2, W3, w1t, w2t, w3t);

    const int B = in_sizes[0] / 784;                            // 65536
    const size_t lds_bytes = (size_t)(MT * XAS + 2 * MT * HS) * sizeof(__bf16); // 138240
    fused_net<<<B / MT, 256, lds_bytes, stream>>>(
        x, b1, b2, b3, w1t, w2t, w3t, out);
}
